// GCL_65970697666596
// MI455X (gfx1250) — compile-verified, matrix operand order checked by reading, not run on hardware
//
#include <hip/hip_runtime.h>
#include <hip/hip_bf16.h>
#include <stdint.h>

typedef __bf16 bf16_t;
typedef __attribute__((ext_vector_type(16))) __bf16 v16bf;
typedef __attribute__((ext_vector_type(8)))  __bf16 v8bf;
typedef __attribute__((ext_vector_type(8)))  float  v8f;

__device__ __forceinline__ void atomAdd(float* p, float v) {
  unsafeAtomicAdd(p, v);   // lowers to global_atomic_add_f32
}

// Pin the per-tile issue pattern: 2 chunks of B loads ahead, then {WMMA, 2 DS reads}.
// 0x100 = DS-read group, 0x008 = MFMA/WMMA group.
template <int KCHUNKS>
__device__ __forceinline__ void sched_tile_pipeline() {
#if defined(__has_builtin)
#if __has_builtin(__builtin_amdgcn_sched_group_barrier)
  __builtin_amdgcn_sched_group_barrier(0x100, 4, 0);   // chunks 0,1 B loads up front
#pragma unroll
  for (int i = 0; i < KCHUNKS - 2; ++i) {
    __builtin_amdgcn_sched_group_barrier(0x008, 1, 0); // WMMA on chunk i
    __builtin_amdgcn_sched_group_barrier(0x100, 2, 0); // loads for chunk i+2
  }
  __builtin_amdgcn_sched_group_barrier(0x008, 2, 0);   // drain last 2 WMMAs
#endif
#endif
}

// ---------------- degree / symmetric normalization ----------------
__global__ void k_deg_init(float* __restrict__ deg, int N) {
  int i = blockIdx.x * blockDim.x + threadIdx.x;
  if (i < N) deg[i] = 1.0f;                       // self-loop contribution
}
__global__ void k_deg_edges(const int* __restrict__ ecol, const float* __restrict__ ev,
                            float* __restrict__ deg, int E) {
  int e = blockIdx.x * blockDim.x + threadIdx.x;
  if (e < E) atomAdd(&deg[ecol[e]], ev[e]);
}
__global__ void k_dinv(const float* __restrict__ deg, float* __restrict__ dinv, int N) {
  int i = blockIdx.x * blockDim.x + threadIdx.x;
  if (i < N) {
    float d = deg[i];
    dinv[i] = d > 0.0f ? rsqrtf(d) : 0.0f;
  }
}
__global__ void k_nv(const int* __restrict__ erow, const int* __restrict__ ecol,
                     const float* __restrict__ ev, const float* __restrict__ dinv,
                     float* __restrict__ nv, int E) {
  int e = blockIdx.x * blockDim.x + threadIdx.x;
  if (e < E) nv[e] = dinv[ecol[e]] * ev[e] * dinv[erow[e]];
}

// ---------------- aggregation (SpMM) ----------------
// out[i][:] = bias[:] + dinv[i]^2 * h[i][:]   (self-loop term + bias pre-fold)
__global__ void k_agg_init(const float* __restrict__ h, const float* __restrict__ bias,
                           const float* __restrict__ dinv, float* __restrict__ out,
                           int N, int F, int f4sh) {
  int idx = blockIdx.x * blockDim.x + threadIdx.x;
  int total4 = N * (F >> 2);
  if (idx >= total4) return;
  int i  = idx >> f4sh;
  int f4 = idx & ((F >> 2) - 1);
  float di = dinv[i];
  float s  = di * di;
  float4 hv = ((const float4*)(h + (size_t)i * F))[f4];
  float4 bv = ((const float4*)bias)[f4];
  float4 o;
  o.x = bv.x + s * hv.x; o.y = bv.y + s * hv.y;
  o.z = bv.z + s * hv.z; o.w = bv.w + s * hv.w;
  ((float4*)(out + (size_t)i * F))[f4] = o;
}

// out[ecol[e]][:] += nv[e] * h[erow[e]][:]  — F/4 float4 lanes per edge, L2 atomics
__global__ void k_agg_edges(const float* __restrict__ h, const int* __restrict__ erow,
                            const int* __restrict__ ecol, const float* __restrict__ nv,
                            float* __restrict__ out, long long E, int F, int f4sh) {
  long long g = (long long)blockIdx.x * blockDim.x + threadIdx.x;
  if (g >= (E << f4sh)) return;
  int e  = (int)(g >> f4sh);
  int f4 = (int)(g & ((1 << f4sh) - 1));
  float w = nv[e];
  int src = erow[e], dst = ecol[e];
  float4 hv = ((const float4*)(h + (size_t)src * F))[f4];
  float* op = out + (size_t)dst * F + (f4 << 2);
  atomAdd(op + 0, w * hv.x);
  atomAdd(op + 1, w * hv.y);
  atomAdd(op + 2, w * hv.z);
  atomAdd(op + 3, w * hv.w);
}

// ---------------- bf16 WMMA GEMM:  C = op(A) @ W (+biasOut)(+relu) ----------------
// A: [M,K] f32 row-major, W: [K,Nc] f32 row-major, C: [M,Nc] f32 row-major.
// W staged transposed in LDS as bf16 [Nc][K+8]; A fragments live in VGPRs and
// are reused across all N tiles; B loads are pipelined 2 chunks ahead of the
// WMMA chain via sched_group_barrier.
template <int K, int Nc, bool RELU_IN, bool BIAS_OUT, bool RELU_OUT>
__global__ __launch_bounds__(256) void k_gemm_wmma(
    const float* __restrict__ A, const float* __restrict__ W,
    const float* __restrict__ biasOut, float* __restrict__ C, int M)
{
  extern __shared__ char smem_raw[];
  bf16_t* Wt = (bf16_t*)smem_raw;
  constexpr int KSTRIDE = K + 8;               // +8 bf16 pad: dodge LDS bank conflicts
  constexpr int NSH     = (Nc == 256) ? 8 : 7; // log2(Nc)
  constexpr int KCHUNKS = K >> 5;              // K/32
  constexpr int NTILES  = Nc >> 4;             // Nc/16

  for (int idx = threadIdx.x; idx < K * Nc; idx += 256) {
    int k = idx >> NSH;
    int n = idx & (Nc - 1);
    Wt[n * KSTRIDE + k] = (bf16_t)W[idx];      // transpose + fp32->bf16
  }
  __syncthreads();

  const int wave  = threadIdx.x >> 5;
  const int lane  = threadIdx.x & 31;
  const int mbase = (blockIdx.x * 8 + wave) * 16;
  if (mbase >= M) return;

  const int lrow  = lane & 15;
  const int khalf = (lane >> 4) << 3;          // 0 or 8 (16-bit A VGPR layout)
  int m = mbase + lrow;
  if (m >= M) m = M - 1;                       // clamp (stores masked below)
  const float* arow = A + (size_t)m * K;

  // A fragments for the whole K extent, kept in registers.
  v16bf afrag[KCHUNKS];
#pragma unroll
  for (int kc = 0; kc < KCHUNKS; ++kc) {
    const float* p0 = arow + kc * 32 + khalf;  // elems 0..7  -> K = base+khalf+j
    const float* p1 = p0 + 16;                 // elems 8..15 -> K = base+16+khalf+j
#pragma unroll
    for (int j = 0; j < 8; ++j) {
      float x0 = p0[j], x1 = p1[j];
      if (RELU_IN) { x0 = fmaxf(x0, 0.0f); x1 = fmaxf(x1, 0.0f); }
      afrag[kc][j]     = (bf16_t)x0;
      afrag[kc][j + 8] = (bf16_t)x1;
    }
  }

  // Per-tile output bias, preloaded (only instantiated when BIAS_OUT).
  float bo[NTILES];
  if (BIAS_OUT) {
#pragma unroll
    for (int nt = 0; nt < NTILES; ++nt) bo[nt] = biasOut[(nt << 4) + lrow];
  }

  // Compute one 16x16 tile with a pinned DS-read/WMMA issue pattern.
  auto computeTile = [&](int nt) -> v8f {
    const int n = (nt << 4) + lrow;
    const bf16_t* wrow = Wt + n * KSTRIDE;
    v16bf bfr[KCHUNKS];
#pragma unroll
    for (int kc = 0; kc < KCHUNKS; ++kc) {
      const int k0 = kc * 32 + khalf;
      v8bf lo = *(const v8bf*)(wrow + k0);          // ds_load_b128
      v8bf hi = *(const v8bf*)(wrow + k0 + 16);     // ds_load_b128
      bfr[kc] = __builtin_shufflevector(lo, hi,
          0, 1, 2, 3, 4, 5, 6, 7, 8, 9, 10, 11, 12, 13, 14, 15);
    }
    v8f acc = {0.f, 0.f, 0.f, 0.f, 0.f, 0.f, 0.f, 0.f};
#pragma unroll
    for (int kc = 0; kc < KCHUNKS; ++kc) {
      acc = __builtin_amdgcn_wmma_f32_16x16x32_bf16(
          false, afrag[kc], false, bfr[kc], (short)0, acc, false, false);
    }
    sched_tile_pipeline<KCHUNKS>();
    return acc;
  };

  const int rbase = mbase + khalf;             // C/D layout: lanes 16-31 hold M = r+8
  // Wave-uniform scalar branch: almost always the full-tile fast path.
  if (__builtin_amdgcn_readfirstlane(mbase + 16 <= M ? 1 : 0)) {
    for (int nt = 0; nt < NTILES; ++nt) {
      v8f acc = computeTile(nt);
      float* crow = C + (size_t)rbase * Nc + ((nt << 4) + lrow);
#pragma unroll
      for (int r = 0; r < 8; ++r) {
        float v = acc[r];
        if (BIAS_OUT) v += bo[nt];
        if (RELU_OUT) v = fmaxf(v, 0.0f);
        crow[(size_t)r * Nc] = v;               // lanes 0..15 contiguous in n
      }
    }
  } else {
    for (int nt = 0; nt < NTILES; ++nt) {
      v8f acc = computeTile(nt);
      float* crow = C + (size_t)rbase * Nc + ((nt << 4) + lrow);
#pragma unroll
      for (int r = 0; r < 8; ++r) {
        if (rbase + r < M) {
          float v = acc[r];
          if (BIAS_OUT) v += bo[nt];
          if (RELU_OUT) v = fmaxf(v, 0.0f);
          crow[(size_t)r * Nc] = v;
        }
      }
    }
  }
}

template <int K, int Nc, bool RELU_IN, bool BIAS_OUT, bool RELU_OUT>
static void launch_gemm(const float* A, const float* W, const float* bo, float* C,
                        int M, hipStream_t stream) {
  size_t lds = (size_t)Nc * (size_t)(K + 8) * sizeof(bf16_t);
  int gx = (M + 127) / 128;
  k_gemm_wmma<K, Nc, RELU_IN, BIAS_OUT, RELU_OUT><<<gx, 256, lds, stream>>>(A, W, bo, C, M);
}

extern "C" void kernel_launch(void* const* d_in, const int* in_sizes, int n_in,
                              void* d_out, int out_size, void* d_ws, size_t ws_size,
                              hipStream_t stream) {
  (void)n_in; (void)out_size; (void)ws_size;
  const float* x    = (const float*)d_in[0];
  const int*   erow = (const int*)  d_in[1];
  const int*   ecol = (const int*)  d_in[2];
  const float* ev   = (const float*)d_in[3];
  const float* W1   = (const float*)d_in[4];
  const float* b1   = (const float*)d_in[5];
  const float* W2   = (const float*)d_in[6];
  const float* b2   = (const float*)d_in[7];
  const float* W3   = (const float*)d_in[8];
  const float* b3   = (const float*)d_in[9];
  const float* P1   = (const float*)d_in[10];
  const float* pb1  = (const float*)d_in[11];
  const float* P2   = (const float*)d_in[12];
  const float* pb2  = (const float*)d_in[13];

  const int IN = 256, HID = 256, EMB = 128, PROJ = 128;
  const int N = in_sizes[0] / IN;
  const long long E = in_sizes[1];

  float* out = (float*)d_out;
  float* emb = out;                       // [N, EMB]
  float* z   = out + (size_t)N * EMB;     // [N, PROJ]

  float* deg  = (float*)d_ws;
  float* dinv = deg + N;
  float* nv   = dinv + N;
  float* B1   = (float*)(((uintptr_t)(nv + E) + 255) & ~(uintptr_t)255);
  float* B2   = B1 + (size_t)N * HID;

  const int T = 256;
  k_deg_init <<<(N + T - 1) / T, T, 0, stream>>>(deg, N);
  k_deg_edges<<<((int)E + T - 1) / T, T, 0, stream>>>(ecol, ev, deg, (int)E);
  k_dinv     <<<(N + T - 1) / T, T, 0, stream>>>(deg, dinv, N);
  k_nv       <<<((int)E + T - 1) / T, T, 0, stream>>>(erow, ecol, ev, dinv, nv, (int)E);

  auto agg = [&](const float* h, const float* bias, float* o, int F) {
    int f4sh = (F == 256) ? 6 : 5;        // log2(F/4)
    int total4 = N * (F >> 2);
    k_agg_init<<<(total4 + T - 1) / T, T, 0, stream>>>(h, bias, dinv, o, N, F, f4sh);
    long long tot = E << f4sh;
    k_agg_edges<<<(int)((tot + T - 1) / T), T, 0, stream>>>(h, erow, ecol, nv, o, E, F, f4sh);
  };

  // g1 = x @ W1
  launch_gemm<256, 256, false, false, false>(x,  W1, nullptr, B1, N, stream);
  agg(B1, b1, B2, HID);                        // B2 = agg(g1) + b1
  // g2 = relu(B2) @ W2
  launch_gemm<256, 256, true,  false, false>(B2, W2, nullptr, B1, N, stream);
  agg(B1, b2, B2, HID);                        // B2 = agg(g2) + b2
  // g3 = relu(B2) @ W3
  launch_gemm<256, 128, true,  false, false>(B2, W3, nullptr, B1, N, stream);
  agg(B1, b3, emb, EMB);                       // emb = agg(g3) + b3  (output 1)
  // t = relu(emb @ P1 + pb1)
  launch_gemm<128, 128, false, true,  true >(emb, P1, pb1, B2, N, stream);
  // z = t @ P2 + pb2  (output 2)
  launch_gemm<128, 128, false, true,  false>(B2, P2, pb2, z, N, stream);
}
